// Block_72816875536638
// MI455X (gfx1250) — matrix operand order repaired and checked by reading, not compiled, not run on hardware
//
#include <hip/hip_runtime.h>
#include <hip/hip_bf16.h>
#include <math.h>

// ---------------- problem constants ----------------
#define Bn 2
#define Tn 2048
#define Cn 512
#define Hn 8
#define Dn 64
#define Mrows (Bn * Tn)   // 4096

typedef __attribute__((ext_vector_type(16))) _Float16 v16h;
typedef __attribute__((ext_vector_type(8)))  _Float16 v8h;
typedef __attribute__((ext_vector_type(8)))  float    v8f;
typedef __attribute__((ext_vector_type(4)))  int      v4i;

// ---------------------------------------------------------------------------
// Async global->LDS staging (gfx1250 GLOBAL_LOAD_ASYNC_TO_LDS_B128, ASYNCcnt)
// Builtin signature (from compiler diagnostic): (v4i addrspace(1)*, v4i addrspace(3)*, imm, imm)
// with a synchronous fallback if the toolchain lacks the builtin.
// ---------------------------------------------------------------------------
#if defined(__gfx1250__) && __has_builtin(__builtin_amdgcn_global_load_async_to_lds_b128)
#define ASYNC_COPY16(dst_lds, src_glob)                                        \
  __builtin_amdgcn_global_load_async_to_lds_b128(                              \
      (__attribute__((address_space(1))) v4i*)(src_glob),                      \
      (__attribute__((address_space(3))) v4i*)(dst_lds), 0, 0)
#if __has_builtin(__builtin_amdgcn_s_wait_asynccnt)
#define ASYNC_WAIT() __builtin_amdgcn_s_wait_asynccnt(0)
#else
#define ASYNC_WAIT() asm volatile("s_wait_asynccnt 0x0" ::: "memory")
#endif
#else
#define ASYNC_COPY16(dst_lds, src_glob) (*(v8h*)(dst_lds) = *(const v8h*)(src_glob))
#define ASYNC_WAIT()
#endif

// ---------------------------------------------------------------------------
// WMMA fragment loaders (wave32, 16x16x32 f16 -> f32)
// A 16x32 (MxK): lane<16 -> row M=lane, elems = K {0..7, 16..23}
//                lane>=16 -> row M=lane-16, elems = K {8..15, 24..31}
// B from LDS transposed tile Bt[N][K]: lane -> col N=lane%16,
//                elems = K { hi*8 + 0..7, hi*8 + 16..23 }   (hi = lane/16)
// C/D: lane -> N=lane%16, vgpr r -> M = r + (lane/16)*8
// ---------------------------------------------------------------------------
__device__ __forceinline__ v16h load_frag_a(const _Float16* base, int row0,
                                            int k0, int stride, int lane) {
  int r = lane & 15, hi = lane >> 4;
  const _Float16* p = base + (size_t)(row0 + r) * stride + k0 + hi * 8;
  v8h lo  = *(const v8h*)(p);        // K = hi*8 + 0..7
  v8h hi8 = *(const v8h*)(p + 16);   // K = hi*8 + 16..23
  v16h out;
#pragma unroll
  for (int i = 0; i < 8; ++i) { out[i] = lo[i]; out[i + 8] = hi8[i]; }
  return out;
}

__device__ __forceinline__ v16h load_frag_b(const _Float16* baseT, int n0,
                                            int k0, int stride, int lane) {
  int c = lane & 15, hi = lane >> 4;
  const _Float16* p = baseT + (size_t)(n0 + c) * stride + k0 + hi * 8;
  v8h a = *(const v8h*)(p);
  v8h b = *(const v8h*)(p + 16);
  v16h out;
#pragma unroll
  for (int i = 0; i < 8; ++i) { out[i] = a[i]; out[i + 8] = b[i]; }
  return out;
}

#define WMMA_F16(a, b, c) \
  __builtin_amdgcn_wmma_f32_16x16x32_f16(false, (a), false, (b), (short)0, (c), false, false)

// ---------------------------------------------------------------------------
// f32 -> f16 conversion (weights)
// ---------------------------------------------------------------------------
__global__ void cvt_f32_f16(const float* __restrict__ src,
                            _Float16* __restrict__ dst, int n) {
  int i = blockIdx.x * blockDim.x + threadIdx.x;
  if (i < n) dst[i] = (_Float16)src[i];
}

// ---------------------------------------------------------------------------
// LayerNorm: one wave per row of 512, output f16. block=256 (8 waves/rows)
// ---------------------------------------------------------------------------
__global__ __launch_bounds__(256) void layernorm_f16(
    const float* __restrict__ in, const float* __restrict__ g,
    const float* __restrict__ b, _Float16* __restrict__ out) {
  int lane = threadIdx.x & 31;
  int wv   = threadIdx.x >> 5;
  int row  = blockIdx.x * 8 + wv;
  const float* xr = in + (size_t)row * Cn;
  float vals[16];
  float s = 0.f;
#pragma unroll
  for (int i = 0; i < 16; ++i) { vals[i] = xr[lane + i * 32]; s += vals[i]; }
#pragma unroll
  for (int off = 16; off > 0; off >>= 1) s += __shfl_xor(s, off, 32);
  float mean = s * (1.0f / Cn);
  float sq = 0.f;
#pragma unroll
  for (int i = 0; i < 16; ++i) { float d = vals[i] - mean; sq += d * d; }
#pragma unroll
  for (int off = 16; off > 0; off >>= 1) sq += __shfl_xor(sq, off, 32);
  float inv = rsqrtf(sq * (1.0f / Cn) + 1e-5f);
  _Float16* orow = out + (size_t)row * Cn;
#pragma unroll
  for (int i = 0; i < 16; ++i) {
    int c = lane + i * 32;
    orow[c] = (_Float16)((vals[i] - mean) * inv * g[c] + b[c]);
  }
}

// ---------------------------------------------------------------------------
// WMMA GEMM: C[M,N] = act(A[M,K]@W[K,N] + bias [+ addend])
// block = 256 threads (8 waves), tile 128x128, K-step 32.
// wave -> 32(M) x 64(N): acc[2][4], 6 fragment loads per 8 WMMAs.
// ---------------------------------------------------------------------------
template <int ACT, bool HAS_ADD, bool HAS_OUTF, bool HAS_OUTH>
__global__ __launch_bounds__(256) void gemm_wmma(
    const _Float16* __restrict__ A, const _Float16* __restrict__ W,
    const float* __restrict__ bias, const float* __restrict__ addend,
    float* __restrict__ outF, _Float16* __restrict__ outH,
    int M, int N, int K) {
  __shared__ __align__(16) _Float16 As[128 * 32];   // [m][k]
  __shared__ __align__(16) _Float16 Bt[128 * 32];   // [n][k] (transposed W tile)
  int tid  = threadIdx.x;
  int lane = tid & 31;
  int wave = tid >> 5;
  int wr = (wave >> 1) * 32;       // wave row offset within 128x128 tile
  int wc = (wave & 1) * 64;        // wave col offset
  int m0 = blockIdx.y * 128;
  int n0 = blockIdx.x * 128;

  v8f acc[2][4] = {};

  for (int k0 = 0; k0 < K; k0 += 32) {
    // stage A tile 128x32 (async b128: 2 chunks of 16B per thread)
#pragma unroll
    for (int rep = 0; rep < 2; ++rep) {
      int c = rep * 256 + tid;             // chunk id, 512 chunks total
      int r = c >> 2, col8 = (c & 3) * 8;
      ASYNC_COPY16(&As[r * 32 + col8], A + (size_t)(m0 + r) * K + k0 + col8);
    }
    // stage W tile transposed: Bt[n][k] = W[k0+k][n0+n]
#pragma unroll
    for (int rep = 0; rep < 16; ++rep) {
      int idx = rep * 256 + tid;
      int kk = idx >> 7, nn = idx & 127;
      Bt[nn * 32 + kk] = W[(size_t)(k0 + kk) * N + n0 + nn];
    }
    ASYNC_WAIT();
    __syncthreads();

    v16h a0 = load_frag_a(As, wr + 0, 0, 32, lane);
    v16h a1 = load_frag_a(As, wr + 16, 0, 32, lane);
    v16h b[4];
#pragma unroll
    for (int nt = 0; nt < 4; ++nt)
      b[nt] = load_frag_b(Bt, wc + nt * 16, 0, 32, lane);
#pragma unroll
    for (int nt = 0; nt < 4; ++nt) {
      acc[0][nt] = WMMA_F16(a0, b[nt], acc[0][nt]);
      acc[1][nt] = WMMA_F16(a1, b[nt], acc[1][nt]);
    }
    __syncthreads();
  }

  int r_ = lane & 15, hi = lane >> 4;
#pragma unroll
  for (int mt = 0; mt < 2; ++mt)
#pragma unroll
    for (int nt = 0; nt < 4; ++nt) {
      int n = n0 + wc + nt * 16 + r_;
      float bn = bias[n];
#pragma unroll
      for (int e = 0; e < 8; ++e) {
        int m = m0 + wr + mt * 16 + hi * 8 + e;
        size_t idx = (size_t)m * N + n;
        float v = acc[mt][nt][e] + bn;
        if (HAS_ADD) v += addend[idx];
        if (ACT == 1) v = 0.5f * v * (1.0f + erff(v * 0.70710678118f));
        if (HAS_OUTF) outF[idx] = v;
        if (HAS_OUTH) outH[idx] = (_Float16)v;
      }
    }
}

// ---------------------------------------------------------------------------
// Split qkv [B*T, 3C] -> q,k,v each [B,H,T,D]
// ---------------------------------------------------------------------------
__global__ void split_qkv(const _Float16* __restrict__ qkv,
                          _Float16* __restrict__ q, _Float16* __restrict__ k,
                          _Float16* __restrict__ v) {
  int i = blockIdx.x * blockDim.x + threadIdx.x;
  int total = Mrows * 3 * Cn;
  if (i >= total) return;
  int col = i % (3 * Cn);
  int bt  = i / (3 * Cn);
  int which = col / Cn;
  int h = (col % Cn) / Dn;
  int d = col % Dn;
  int b = bt / Tn, t = bt % Tn;
  size_t dst = (((size_t)(b * Hn + h)) * Tn + t) * Dn + d;
  _Float16 val = qkv[i];
  if (which == 0) q[dst] = val;
  else if (which == 1) k[dst] = val;
  else v[dst] = val;
}

// ---------------------------------------------------------------------------
// Attention: per (b,h), 64-row blocks. Two WMMA stages per 64-wide K/V tile:
//   S = Q K^T ;  P = sigmoid(S/8) * coulomb ;  Y += P V
// block = 128 threads (4 waves). Output to yatt[B*T, C] (bt, h*64+d) f32.
// ---------------------------------------------------------------------------
__global__ __launch_bounds__(128) void attention_wmma(
    const _Float16* __restrict__ q, const _Float16* __restrict__ k,
    const _Float16* __restrict__ v, const float* __restrict__ coulomb,
    float* __restrict__ yatt) {
  __shared__ __align__(16) _Float16 Qs[64 * 64];   // [r][d]
  __shared__ __align__(16) _Float16 Ks[64 * 64];   // [s][d]  == Bt layout for stage 1
  __shared__ __align__(16) _Float16 Vt[64 * 64];   // [d][s]  == Bt layout for stage 2
  __shared__ __align__(16) _Float16 Ps[64 * 64];   // [r][s]

  int tid  = threadIdx.x;
  int lane = tid & 31;
  int wave = tid >> 5;
  int wr = (wave >> 1) * 32;
  int wc = (wave & 1) * 32;
  int r_ = lane & 15, hi = lane >> 4;

  int bh   = blockIdx.y;            // b*H + h
  int row0 = blockIdx.x * 64;
  int b    = bh / Hn;
  int head = bh % Hn;
  const float scale = 0.125f;       // 1/sqrt(64)
  const float* cm = coulomb + (size_t)b * Tn * Tn;

  // load Q block (64x64 halves, contiguous) via async copies
  const _Float16* qg = q + ((size_t)bh * Tn + row0) * Dn;
#pragma unroll
  for (int rep = 0; rep < 4; ++rep) {
    int c = rep * 128 + tid;
    ASYNC_COPY16(&Qs[c * 8], qg + c * 8);
  }

  v8f yacc[2][2] = {};

  for (int s0 = 0; s0 < Tn; s0 += 64) {
    // stage K tile (row-major = [s][d], directly usable as Bt for S=Q K^T)
    const _Float16* kg = k + ((size_t)bh * Tn + s0) * Dn;
#pragma unroll
    for (int rep = 0; rep < 4; ++rep) {
      int c = rep * 128 + tid;
      ASYNC_COPY16(&Ks[c * 8], kg + c * 8);
    }
    // stage V tile transposed: Vt[d][s] = V[s0+s][d]
    const _Float16* vg = v + ((size_t)bh * Tn + s0) * Dn;
#pragma unroll
    for (int rep = 0; rep < 32; ++rep) {
      int idx = rep * 128 + tid;
      int s = idx >> 6, d = idx & 63;
      Vt[d * 64 + s] = vg[(size_t)s * Dn + d];
    }
    ASYNC_WAIT();
    __syncthreads();

    // stage 1: S = Q @ K^T  (K dim = d, 2 steps of 32)
    v8f sacc[2][2] = {};
#pragma unroll
    for (int kd = 0; kd < 64; kd += 32) {
      v16h a0 = load_frag_a(Qs, wr + 0, kd, 64, lane);
      v16h a1 = load_frag_a(Qs, wr + 16, kd, 64, lane);
      v16h b0 = load_frag_b(Ks, wc + 0, kd, 64, lane);
      v16h b1 = load_frag_b(Ks, wc + 16, kd, 64, lane);
      sacc[0][0] = WMMA_F16(a0, b0, sacc[0][0]);
      sacc[0][1] = WMMA_F16(a0, b1, sacc[0][1]);
      sacc[1][0] = WMMA_F16(a1, b0, sacc[1][0]);
      sacc[1][1] = WMMA_F16(a1, b1, sacc[1][1]);
    }

    // epilogue: P = sigmoid(S*scale) * coulomb  -> LDS (f16)
#pragma unroll
    for (int mt = 0; mt < 2; ++mt)
#pragma unroll
      for (int nt = 0; nt < 2; ++nt)
#pragma unroll
        for (int e = 0; e < 8; ++e) {
          int m = wr + mt * 16 + hi * 8 + e;
          int s = wc + nt * 16 + r_;
          float x = sacc[mt][nt][e] * scale;
          float sig = 1.0f / (1.0f + __expf(-x));
          float p = sig * cm[(size_t)(row0 + m) * Tn + (s0 + s)];
          Ps[m * 64 + s] = (_Float16)p;
        }
    __syncthreads();

    // stage 2: Y += P @ V  (K dim = s, 2 steps of 32)
#pragma unroll
    for (int ks = 0; ks < 64; ks += 32) {
      v16h a0 = load_frag_a(Ps, wr + 0, ks, 64, lane);
      v16h a1 = load_frag_a(Ps, wr + 16, ks, 64, lane);
      v16h b0 = load_frag_b(Vt, wc + 0, ks, 64, lane);
      v16h b1 = load_frag_b(Vt, wc + 16, ks, 64, lane);
      yacc[0][0] = WMMA_F16(a0, b0, yacc[0][0]);
      yacc[0][1] = WMMA_F16(a0, b1, yacc[0][1]);
      yacc[1][0] = WMMA_F16(a1, b0, yacc[1][0]);
      yacc[1][1] = WMMA_F16(a1, b1, yacc[1][1]);
    }
    __syncthreads();
  }

  // write Y block to yatt[(b*T + t), head*64 + d]
#pragma unroll
  for (int mt = 0; mt < 2; ++mt)
#pragma unroll
    for (int nt = 0; nt < 2; ++nt)
#pragma unroll
      for (int e = 0; e < 8; ++e) {
        int m = wr + mt * 16 + hi * 8 + e;
        int d = wc + nt * 16 + r_;
        yatt[(size_t)(b * Tn + row0 + m) * Cn + head * Dn + d] = yacc[mt][nt][e];
      }
}

// ---------------------------------------------------------------------------
// host launcher
// ---------------------------------------------------------------------------
extern "C" void kernel_launch(void* const* d_in, const int* in_sizes, int n_in,
                              void* d_out, int out_size, void* d_ws, size_t ws_size,
                              hipStream_t stream) {
  (void)in_sizes; (void)n_in; (void)out_size; (void)ws_size;
  const float* x        = (const float*)d_in[0];
  const float* coulomb  = (const float*)d_in[2];
  const float* ln1_g    = (const float*)d_in[3];
  const float* ln1_b    = (const float*)d_in[4];
  const float* w_attn   = (const float*)d_in[5];
  const float* b_attn   = (const float*)d_in[6];
  const float* w_self   = (const float*)d_in[7];
  const float* b_self   = (const float*)d_in[8];
  const float* w_proj   = (const float*)d_in[9];
  const float* b_proj   = (const float*)d_in[10];
  const float* ln2_g    = (const float*)d_in[11];
  const float* ln2_b    = (const float*)d_in[12];
  const float* w_fc     = (const float*)d_in[13];
  const float* b_fc     = (const float*)d_in[14];
  const float* w_fcp    = (const float*)d_in[15];
  const float* b_fcp    = (const float*)d_in[16];

  char* ws = (char*)d_ws;
  size_t off = 0;
  auto alloc = [&](size_t bytes) -> void* {
    void* p = ws + off;
    off += (bytes + 255) & ~(size_t)255;
    return p;
  };

  _Float16* wattn16 = (_Float16*)alloc((size_t)Cn * 3 * Cn * 2);
  _Float16* wself16 = (_Float16*)alloc((size_t)Cn * Cn * 2);
  _Float16* wproj16 = (_Float16*)alloc((size_t)Cn * Cn * 2);
  _Float16* wfc16   = (_Float16*)alloc((size_t)Cn * 4 * Cn * 2);
  _Float16* wfcp16  = (_Float16*)alloc((size_t)4 * Cn * Cn * 2);
  _Float16* h16     = (_Float16*)alloc((size_t)Mrows * Cn * 2);
  _Float16* qkv16   = (_Float16*)alloc((size_t)Mrows * 3 * Cn * 2);
  _Float16* qh      = (_Float16*)alloc((size_t)Bn * Hn * Tn * Dn * 2);
  _Float16* kh      = (_Float16*)alloc((size_t)Bn * Hn * Tn * Dn * 2);
  _Float16* vh      = (_Float16*)alloc((size_t)Bn * Hn * Tn * Dn * 2);
  float*    yatt    = (float*)alloc((size_t)Mrows * Cn * 4);
  _Float16* y1f16   = (_Float16*)alloc((size_t)Mrows * Cn * 2);
  float*    y2f32   = (float*)alloc((size_t)Mrows * Cn * 4);
  _Float16* h2f16   = (_Float16*)alloc((size_t)Mrows * Cn * 2);
  _Float16* g16     = (_Float16*)alloc((size_t)Mrows * 4 * Cn * 2);

  auto cvt = [&](const float* s, _Float16* d, int n) {
    cvt_f32_f16<<<(n + 255) / 256, 256, 0, stream>>>(s, d, n);
  };
  cvt(w_attn, wattn16, Cn * 3 * Cn);
  cvt(w_self, wself16, Cn * Cn);
  cvt(w_proj, wproj16, Cn * Cn);
  cvt(w_fc,   wfc16,   Cn * 4 * Cn);
  cvt(w_fcp,  wfcp16,  4 * Cn * Cn);

  // h = LN1(x)
  layernorm_f16<<<Mrows / 8, 256, 0, stream>>>(x, ln1_g, ln1_b, h16);

  // qkv = h @ w_attn + b_attn
  gemm_wmma<0, false, false, true><<<dim3(3 * Cn / 128, Mrows / 128), 256, 0, stream>>>(
      h16, wattn16, b_attn, nullptr, nullptr, qkv16, Mrows, 3 * Cn, Cn);

  split_qkv<<<(Mrows * 3 * Cn + 255) / 256, 256, 0, stream>>>(qkv16, qh, kh, vh);

  // y_att = (sigmoid(qk^T/8) * coulomb) @ v
  attention_wmma<<<dim3(Tn / 64, Bn * Hn), 128, 0, stream>>>(qh, kh, vh, coulomb, yatt);

  // y1 = y_att + h @ w_self + b_self
  gemm_wmma<0, true, false, true><<<dim3(Cn / 128, Mrows / 128), 256, 0, stream>>>(
      h16, wself16, b_self, yatt, nullptr, y1f16, Mrows, Cn, Cn);

  // y2 = y1 @ w_proj + b_proj
  gemm_wmma<0, false, true, false><<<dim3(Cn / 128, Mrows / 128), 256, 0, stream>>>(
      y1f16, wproj16, b_proj, nullptr, y2f32, nullptr, Mrows, Cn, Cn);

  // h2 = LN2(y2)
  layernorm_f16<<<Mrows / 8, 256, 0, stream>>>(y2f32, ln2_g, ln2_b, h2f16);

  // g = gelu(h2 @ w_fc + b_fc)
  gemm_wmma<1, false, false, true><<<dim3(4 * Cn / 128, Mrows / 128), 256, 0, stream>>>(
      h2f16, wfc16, b_fc, nullptr, nullptr, g16, Mrows, 4 * Cn, Cn);

  // out = g @ w_fc_proj + b_fc_proj
  gemm_wmma<0, false, true, false><<<dim3(Cn / 128, Mrows / 128), 256, 0, stream>>>(
      g16, wfcp16, b_fcp, nullptr, (float*)d_out, nullptr, Mrows, Cn, 4 * Cn);
}
